// MultiheadAttention_34411277976035
// MI455X (gfx1250) — matrix operand order repaired and checked by reading, not compiled, not run on hardware
//
#include <hip/hip_runtime.h>
#include <hip/hip_bf16.h>
#include <math.h>

// ---------------------------------------------------------------------------
// Multihead attention forward, CDNA5 (gfx1250), wave32 WMMA f16 (f32 accum).
// B=2, L=4096, D=512, H=8, HD=64.
// Pipeline: cvt -> QKV GEMM (scatter Q,K,[V^T]) -> flash attention -> out GEMM
// ---------------------------------------------------------------------------

typedef __attribute__((ext_vector_type(16))) _Float16 v16h;
typedef __attribute__((ext_vector_type(8)))  _Float16 v8h;
typedef __attribute__((ext_vector_type(8)))  float    v8f;

constexpr int BATCH  = 2;
constexpr int L_SEQ  = 4096;
constexpr int D_MOD  = 512;
constexpr int NHEAD  = 8;
constexpr int HDIM   = 64;
constexpr int M_ROWS = BATCH * L_SEQ;     // 8192
constexpr int N_QKV  = 3 * D_MOD;         // 1536

// workspace layout (bytes)
constexpr size_t OFF_X16   = 0;
constexpr size_t SZ_X16    = (size_t)M_ROWS * D_MOD * 2;            // 8 MiB
constexpr size_t OFF_WQKVT = OFF_X16 + SZ_X16;
constexpr size_t SZ_WQKVT  = (size_t)N_QKV * D_MOD * 2;             // 1.5 MiB
constexpr size_t OFF_WOT   = OFF_WQKVT + SZ_WQKVT;
constexpr size_t SZ_WOT    = (size_t)D_MOD * D_MOD * 2;             // 0.5 MiB
constexpr size_t OFF_Q16   = OFF_WOT + SZ_WOT;
constexpr size_t SZ_HEADS  = (size_t)BATCH * NHEAD * L_SEQ * HDIM * 2; // 8 MiB
constexpr size_t OFF_K16   = OFF_Q16 + SZ_HEADS;
constexpr size_t OFF_VT16  = OFF_K16 + SZ_HEADS;
constexpr size_t OFF_AO16  = OFF_VT16 + SZ_HEADS;                   // attn out f16

// ---------------------------------------------------------------------------
// WMMA helpers
// ---------------------------------------------------------------------------
__device__ inline v8f v8f_zero() {
    v8f z;
#pragma unroll
    for (int i = 0; i < 8; ++i) z[i] = 0.0f;
    return z;
}

__device__ inline v8f wmma_f16(v16h a, v16h b, v8f c) {
    return __builtin_amdgcn_wmma_f32_16x16x32_f16(
        /*neg_a=*/false, a, /*neg_b=*/false, b,
        /*c_mod=*/(short)0, c, /*reuse_a=*/false, /*reuse_b=*/false);
}

// A fragment (16x32 f16) from row-major [m][k], lda in halfs.
// CDNA5 layout: lane: m = lane&15, half = lane>>4;
//   e=0..7  -> k = kbase + 8*half + e
//   e=8..15 -> k = kbase + 16 + 8*half + (e-8)
__device__ inline v16h load_a_frag(const _Float16* __restrict__ A, int lda,
                                   int mbase, int kbase) {
    const int lane = threadIdx.x & 31;
    const int m    = mbase + (lane & 15);
    const int hf   = lane >> 4;
    const _Float16* p = A + (size_t)m * lda + kbase + 8 * hf;
    v8h lo = *(const v8h*)(p);
    v8h hi = *(const v8h*)(p + 16);
    v16h r;
#pragma unroll
    for (int i = 0; i < 8; ++i) { r[i] = lo[i]; r[8 + i] = hi[i]; }
    return r;
}

// B fragment (32x16 f16) from *transposed* storage BT[n][k], ldb in halfs.
// Layout: lane: n = lane&15, half = lane>>4; e=0..7 -> k = kbase + 8*half + e,
//   e=8..15 -> k = kbase + 16 + 8*half + (e-8).
__device__ inline v16h load_b_frag(const _Float16* __restrict__ BT, int ldb,
                                   int nbase, int kbase) {
    const int lane = threadIdx.x & 31;
    const int n    = nbase + (lane & 15);
    const int hf   = lane >> 4;
    const _Float16* p = BT + (size_t)n * ldb + kbase + 8 * hf;
    v8h lo = *(const v8h*)(p);
    v8h hi = *(const v8h*)(p + 16);
    v16h r;
#pragma unroll
    for (int i = 0; i < 8; ++i) { r[i] = lo[i]; r[8 + i] = hi[i]; }
    return r;
}

// ---------------------------------------------------------------------------
// Conversions
// ---------------------------------------------------------------------------
__global__ __launch_bounds__(256) void cvt_f32_f16(const float* __restrict__ in,
                                                   _Float16* __restrict__ out,
                                                   int n) {
    int i = blockIdx.x * 256 + threadIdx.x;
    if (i < n) out[i] = (_Float16)in[i];
}

// in [K][N] f32 -> out [N][K] f16
__global__ __launch_bounds__(256) void cvt_transpose(const float* __restrict__ in,
                                                     _Float16* __restrict__ out,
                                                     int K, int N) {
    int i = blockIdx.x * 256 + threadIdx.x;
    if (i < K * N) {
        int k = i / N, n = i - k * N;
        out[(size_t)n * K + k] = (_Float16)in[i];
    }
}

// ---------------------------------------------------------------------------
// GEMM 1: qkv = x @ w_qkv + b_qkv ; scatter to Q/K [B,H,L,64], V^T [B,H,64,L]
// A: x16 [8192][512] f16 row-major.  BT: wqkvT [1536][512] f16.
// One wave computes a 32(M) x 64(N) tile (2 A frags, 8 accumulators).
// ---------------------------------------------------------------------------
__global__ __launch_bounds__(256) void gemm_qkv_kernel(
    const _Float16* __restrict__ A, const _Float16* __restrict__ BT,
    const float* __restrict__ bias, _Float16* __restrict__ q,
    _Float16* __restrict__ k, _Float16* __restrict__ vT) {
    const int lane = threadIdx.x & 31;
    const int wave = threadIdx.x >> 5;
    const int tile = blockIdx.x * 8 + wave;
    constexpr int NT = N_QKV / 64;   // 24
    const int mt = tile / NT, nt = tile - mt * NT;
    const int mbase = mt * 32, nbase = nt * 64;

    v8f acc[2][4];
#pragma unroll
    for (int r = 0; r < 2; ++r)
#pragma unroll
        for (int c = 0; c < 4; ++c) acc[r][c] = v8f_zero();

    for (int ks = 0; ks < D_MOD; ks += 32) {
        v16h a0 = load_a_frag(A, D_MOD, mbase, ks);
        v16h a1 = load_a_frag(A, D_MOD, mbase + 16, ks);
#pragma unroll
        for (int c = 0; c < 4; ++c) {
            v16h b = load_b_frag(BT, D_MOD, nbase + 16 * c, ks);
            acc[0][c] = wmma_f16(a0, b, acc[0][c]);
            acc[1][c] = wmma_f16(a1, b, acc[1][c]);
        }
    }

    // epilogue: C layout -> lane: n = lane&15, row m = v + 8*(lane>>4)
    const int ml = lane & 15;
    const int hf = lane >> 4;
#pragma unroll
    for (int r = 0; r < 2; ++r) {
#pragma unroll
        for (int c = 0; c < 4; ++c) {
#pragma unroll
            for (int v = 0; v < 8; ++v) {
                int m   = mbase + 16 * r + v + 8 * hf;   // row in [0, 8192)
                int col = nbase + c * 16 + ml;           // col in [0, 1536)
                float val = acc[r][c][v] + bias[col];
                int bb = m >> 12;                        // / L_SEQ
                int l  = m & (L_SEQ - 1);
                int h  = col / 192;
                int rr = col - 192 * h;
                int kind = rr >> 6;                      // 0=q 1=k 2=v
                int d  = rr & 63;
                if (kind == 0) {
                    q[(((size_t)bb * NHEAD + h) * L_SEQ + l) * HDIM + d] = (_Float16)val;
                } else if (kind == 1) {
                    k[(((size_t)bb * NHEAD + h) * L_SEQ + l) * HDIM + d] = (_Float16)val;
                } else {
                    vT[(((size_t)bb * NHEAD + h) * HDIM + d) * L_SEQ + l] = (_Float16)val;
                }
            }
        }
    }
}

// ---------------------------------------------------------------------------
// Flash attention: one wave per (b, h, 32-row Q block). Online softmax.
// K/V fragments shared across the two 16-row halves (2x operand reuse).
// ---------------------------------------------------------------------------
__global__ __launch_bounds__(256) void attention_kernel(
    const _Float16* __restrict__ q, const _Float16* __restrict__ k,
    const _Float16* __restrict__ vT, _Float16* __restrict__ ao) {
    __shared__ _Float16 p_lds[8][32][32];    // per-wave P transpose tile (16 KiB)

    const int lane = threadIdx.x & 31;
    const int wave = threadIdx.x >> 5;
    const int tile = blockIdx.x * 8 + wave;              // 0 .. 2047
    const int bb  = tile >> 10;                          // / (H * L/32)
    const int rem = tile & 1023;
    const int h   = rem >> 7;
    const int qb  = rem & 127;

    const int ml = lane & 15;
    const int hf = lane >> 4;
    const float scale = 0.125f;                          // 1/sqrt(64)

    const _Float16* qbase  = q  + (((size_t)bb * NHEAD + h) * L_SEQ + (size_t)qb * 32) * HDIM;
    const _Float16* kbase  = k  + (((size_t)bb * NHEAD + h) * L_SEQ) * HDIM;
    const _Float16* vtbase = vT + (((size_t)bb * NHEAD + h) * HDIM) * L_SEQ;

    v16h qa[2][2];
#pragma unroll
    for (int r = 0; r < 2; ++r) {
        qa[r][0] = load_a_frag(qbase, HDIM, 16 * r, 0);
        qa[r][1] = load_a_frag(qbase, HDIM, 16 * r, 32);
    }

    v8f o[2][4];
#pragma unroll
    for (int r = 0; r < 2; ++r)
#pragma unroll
        for (int c = 0; c < 4; ++c) o[r][c] = v8f_zero();
    float m_run[2][8], d_run[2][8];
#pragma unroll
    for (int r = 0; r < 2; ++r)
#pragma unroll
        for (int v = 0; v < 8; ++v) { m_run[r][v] = -INFINITY; d_run[r][v] = 0.0f; }

    for (int jj = 0; jj < L_SEQ; jj += 32) {
        // K fragments for 32 keys, shared across both 16-row halves
        v16h kb0 = load_b_frag(kbase, HDIM, jj, 0);
        v16h kb1 = load_b_frag(kbase, HDIM, jj, 32);
        v16h kb2 = load_b_frag(kbase, HDIM, jj + 16, 0);
        v16h kb3 = load_b_frag(kbase, HDIM, jj + 16, 32);

#pragma unroll
        for (int r = 0; r < 2; ++r) {
            v8f s0 = v8f_zero(), s1 = v8f_zero();
            s0 = wmma_f16(qa[r][0], kb0, s0);
            s0 = wmma_f16(qa[r][1], kb1, s0);
            s1 = wmma_f16(qa[r][0], kb2, s1);
            s1 = wmma_f16(qa[r][1], kb3, s1);

            // online softmax per C-layout row (row = 16r + v + 8*hf, 16 lanes share it)
#pragma unroll
            for (int v = 0; v < 8; ++v) {
                float a0 = s0[v] * scale;
                float a1 = s1[v] * scale;
                float mx = fmaxf(a0, a1);
#pragma unroll
                for (int off = 1; off < 16; off <<= 1)
                    mx = fmaxf(mx, __shfl_xor(mx, off, 32));
                float mnew  = fmaxf(m_run[r][v], mx);
                float alpha = __expf(m_run[r][v] - mnew);
                float p0 = __expf(a0 - mnew);
                float p1 = __expf(a1 - mnew);
                float rs = p0 + p1;
#pragma unroll
                for (int off = 1; off < 16; off <<= 1)
                    rs += __shfl_xor(rs, off, 32);
                d_run[r][v] = d_run[r][v] * alpha + rs;
                m_run[r][v] = mnew;
#pragma unroll
                for (int c = 0; c < 4; ++c) o[r][c][v] *= alpha;
                // C layout -> LDS [row][key-col] for A-layout reload
                p_lds[wave][16 * r + v + 8 * hf][ml]      = (_Float16)p0;
                p_lds[wave][16 * r + v + 8 * hf][16 + ml] = (_Float16)p1;
            }
        }
        asm volatile("s_wait_dscnt 0" ::: "memory");     // same-wave LDS RAW

        // reload P as A fragments (two 16x32 tiles)
        v16h pa[2];
#pragma unroll
        for (int r = 0; r < 2; ++r) {
            const _Float16* prow = &p_lds[wave][16 * r + ml][0];
            v8h lo = *(const v8h*)(prow + 8 * hf);
            v8h hi = *(const v8h*)(prow + 16 + 8 * hf);
#pragma unroll
            for (int i = 0; i < 8; ++i) { pa[r][i] = lo[i]; pa[r][8 + i] = hi[i]; }
        }

        // O += P @ V ; each V fragment feeds both row halves
#pragma unroll
        for (int c = 0; c < 4; ++c) {
            v16h vb = load_b_frag(vtbase, L_SEQ, c * 16, jj);
            o[0][c] = wmma_f16(pa[0], vb, o[0][c]);
            o[1][c] = wmma_f16(pa[1], vb, o[1][c]);
        }
    }

    // normalize and scatter to [B, L, D] f16 (row m, col h*64 + c*16 + n)
#pragma unroll
    for (int r = 0; r < 2; ++r) {
#pragma unroll
        for (int v = 0; v < 8; ++v) {
            float inv = 1.0f / d_run[r][v];
            int m = qb * 32 + 16 * r + v + 8 * hf;
#pragma unroll
            for (int c = 0; c < 4; ++c) {
                int col = h * HDIM + c * 16 + ml;
                ao[((size_t)bb * L_SEQ + m) * D_MOD + col] = (_Float16)(o[r][c][v] * inv);
            }
        }
    }
}

// ---------------------------------------------------------------------------
// GEMM 2: out = attn @ w_o + b_o, f32 output straight to d_out.
// One wave computes a 32(M) x 64(N) tile.
// ---------------------------------------------------------------------------
__global__ __launch_bounds__(256) void gemm_out_kernel(
    const _Float16* __restrict__ A, const _Float16* __restrict__ BT,
    const float* __restrict__ bias, float* __restrict__ out) {
    const int lane = threadIdx.x & 31;
    const int wave = threadIdx.x >> 5;
    const int tile = blockIdx.x * 8 + wave;
    constexpr int NT = D_MOD / 64;   // 8
    const int mt = tile / NT, nt = tile - mt * NT;
    const int mbase = mt * 32, nbase = nt * 64;

    v8f acc[2][4];
#pragma unroll
    for (int r = 0; r < 2; ++r)
#pragma unroll
        for (int c = 0; c < 4; ++c) acc[r][c] = v8f_zero();

    for (int ks = 0; ks < D_MOD; ks += 32) {
        v16h a0 = load_a_frag(A, D_MOD, mbase, ks);
        v16h a1 = load_a_frag(A, D_MOD, mbase + 16, ks);
#pragma unroll
        for (int c = 0; c < 4; ++c) {
            v16h b = load_b_frag(BT, D_MOD, nbase + 16 * c, ks);
            acc[0][c] = wmma_f16(a0, b, acc[0][c]);
            acc[1][c] = wmma_f16(a1, b, acc[1][c]);
        }
    }

    const int ml = lane & 15;
    const int hf = lane >> 4;
#pragma unroll
    for (int r = 0; r < 2; ++r) {
#pragma unroll
        for (int c = 0; c < 4; ++c) {
#pragma unroll
            for (int v = 0; v < 8; ++v) {
                int m   = mbase + 16 * r + v + 8 * hf;
                int col = nbase + c * 16 + ml;
                out[(size_t)m * D_MOD + col] = acc[r][c][v] + bias[col];
            }
        }
    }
}

// ---------------------------------------------------------------------------
// Launch
// ---------------------------------------------------------------------------
extern "C" void kernel_launch(void* const* d_in, const int* in_sizes, int n_in,
                              void* d_out, int out_size, void* d_ws, size_t ws_size,
                              hipStream_t stream) {
    const float* x     = (const float*)d_in[0];
    const float* w_qkv = (const float*)d_in[1];
    const float* b_qkv = (const float*)d_in[2];
    const float* w_o   = (const float*)d_in[3];
    const float* b_o   = (const float*)d_in[4];
    float* out = (float*)d_out;

    char* ws = (char*)d_ws;
    _Float16* x16    = (_Float16*)(ws + OFF_X16);
    _Float16* wqkvT  = (_Float16*)(ws + OFF_WQKVT);
    _Float16* woT    = (_Float16*)(ws + OFF_WOT);
    _Float16* q16    = (_Float16*)(ws + OFF_Q16);
    _Float16* k16    = (_Float16*)(ws + OFF_K16);
    _Float16* vT16   = (_Float16*)(ws + OFF_VT16);
    _Float16* ao16   = (_Float16*)(ws + OFF_AO16);

    (void)in_sizes; (void)n_in; (void)out_size; (void)ws_size;

    // f32 -> f16 conversions (weights transposed for contiguous B fragments)
    cvt_f32_f16<<<(M_ROWS * D_MOD) / 256, 256, 0, stream>>>(x, x16, M_ROWS * D_MOD);
    cvt_transpose<<<(D_MOD * N_QKV) / 256, 256, 0, stream>>>(w_qkv, wqkvT, D_MOD, N_QKV);
    cvt_transpose<<<(D_MOD * D_MOD) / 256, 256, 0, stream>>>(w_o, woT, D_MOD, D_MOD);

    // QKV projection + head scatter: (8192/32)*(1536/64) = 6144 wave-tiles / 8
    gemm_qkv_kernel<<<768, 256, 0, stream>>>(x16, wqkvT, b_qkv, q16, k16, vT16);

    // flash attention: B*H*(L/32) = 2048 wave-tiles / 8
    attention_kernel<<<256, 256, 0, stream>>>(q16, k16, vT16, ao16);

    // output projection: (8192/32)*(512/64) = 2048 wave-tiles / 8
    gemm_out_kernel<<<256, 256, 0, stream>>>(ao16, woT, b_o, out);
}
